// Head_65755949302252
// MI455X (gfx1250) — matrix operand order repaired and checked by reading, not compiled, not run on hardware
//
#include <hip/hip_runtime.h>
#include <hip/hip_bf16.h>

// ---- CDNA5 WMMA types -------------------------------------------------------
typedef __attribute__((ext_vector_type(16))) _Float16 v16h;
typedef __attribute__((ext_vector_type(8)))  _Float16 v8h;
typedef __attribute__((ext_vector_type(4)))  _Float16 v4h;
typedef __attribute__((ext_vector_type(8)))  float    v8f;

union AFrag { v16h v; v8h h8[2]; };

static __device__ __forceinline__ v8f wmma_f16(v16h a, v16h b, v8f c) {
    // D = A(16x32 f16) * B(32x16 f16) + C(16x16 f32)
    return __builtin_amdgcn_wmma_f32_16x16x32_f16(
        /*neg_a=*/false, a, /*neg_b=*/false, b,
        /*c_mod=*/(short)0, c, /*reuse_a=*/false, /*reuse_b=*/false);
}

// LDS byte offset of a generic __shared__ pointer: flat LDS addressing uses
// addr[31:0] as the wave-relative LDS offset, so truncation is exact.
static __device__ __forceinline__ uint32_t lds_off(const void* p) {
    return (uint32_t)(uintptr_t)p;
}

// CDNA5 async copy global -> LDS (16B per lane), tracked by ASYNCcnt.
static __device__ __forceinline__ void async_b128(uint32_t lds, const void* g) {
    asm volatile("global_load_async_to_lds_b128 %0, %1, off"
                 :: "v"(lds), "v"(g) : "memory");
}

// Problem constants
#define Bc 8
#define Tc 2048
#define Cc 1024
#define Hc 64
#define BT (Bc * Tc)   // 16384

// ============================================================================
// Kernel 1: fused QKV projection + RoPE.
//   q,k -> ws row-major [BT,64] f16 ; v -> ws transposed [B,64,T] f16
// Block: 256 threads (8 waves). Each wave: 16 M-rows x 64 N. K loop: 1024/32.
// ============================================================================
__global__ __launch_bounds__(256) void qkv_rope_kernel(
    const float* __restrict__ x,
    const float* __restrict__ Wq, const float* __restrict__ Wk,
    const float* __restrict__ Wv,
    _Float16* __restrict__ qbuf, _Float16* __restrict__ kbuf,
    _Float16* __restrict__ vT)
{
    __shared__ __align__(64) _Float16 sX[128 * 32];      // x tile, row-major
    __shared__ __align__(64) _Float16 sWt[3 * 64 * 32];  // W tiles, transposed [mat][col][k]

    const int tid    = threadIdx.x;
    const int wave   = tid >> 5;
    const int lane   = tid & 31;
    const int lane16 = lane & 15;
    const int g      = lane >> 4;
    const int m0b    = blockIdx.x * 128;        // block row base in [0, BT)
    const int m0     = m0b + wave * 16;         // wave row base

    v8f accq[4] = {}, acck[4] = {}, accv[4] = {};

    for (int ks = 0; ks < Cc; ks += 32) {
        // ---- stage x[128x32] fp32 -> f16: float4 loads, v4h LDS stores ----
#pragma unroll
        for (int it = 0; it < 4; ++it) {
            int idx = tid + it * 256;           // 0..1023 float4 chunks
            int r = idx >> 3, c4 = idx & 7;     // 8 float4 per 32-col row
            float4 w = *(const float4*)&x[(size_t)(m0b + r) * Cc + ks + c4 * 4];
            v4h h;
            h[0] = (_Float16)w.x; h[1] = (_Float16)w.y;
            h[2] = (_Float16)w.z; h[3] = (_Float16)w.w;
            *(v4h*)&sX[r * 32 + c4 * 4] = h;
        }
        // ---- stage Wq/Wk/Wv[32x64] transposed into LDS (float4 loads) ----
#pragma unroll
        for (int it = 0; it < 2; ++it) {
            int idx = tid + it * 256;           // 0..511 float4 chunks
            int kk = idx >> 4, c4 = idx & 15;   // 16 float4 per 64-col row
            size_t go = (size_t)(ks + kk) * Hc + c4 * 4;
            float4 wq = *(const float4*)&Wq[go];
            float4 wk = *(const float4*)&Wk[go];
            float4 wv = *(const float4*)&Wv[go];
            float aq[4] = {wq.x, wq.y, wq.z, wq.w};
            float ak[4] = {wk.x, wk.y, wk.z, wk.w};
            float av[4] = {wv.x, wv.y, wv.z, wv.w};
#pragma unroll
            for (int u = 0; u < 4; ++u) {
                int lo = (c4 * 4 + u) * 32 + kk;
                sWt[lo]            = (_Float16)aq[u];
                sWt[2048 + lo]     = (_Float16)ak[u];
                sWt[2 * 2048 + lo] = (_Float16)av[u];
            }
        }
        __syncthreads();

        if (ks + 32 < Cc)
            __builtin_prefetch(&x[(size_t)(m0b + (tid >> 1)) * Cc + ks + 32], 0, 1);

        // ---- A fragment: row = lane16, K = g*16 + e (matches B) ----
        AFrag a;
        const int rowl = wave * 16 + lane16;
        a.h8[0] = *(const v8h*)&sX[rowl * 32 + g * 16];       // K = g*16 + e
        a.h8[1] = *(const v8h*)&sX[rowl * 32 + g * 16 + 8];   // K = g*16 + 8 + e

        // ---- 12 WMMAs: q,k,v x 4 column tiles ----
#pragma unroll
        for (int nt = 0; nt < 4; ++nt) {
            int col = nt * 16 + lane16;
            v16h bq = *(const v16h*)&sWt[col * 32 + g * 16];            // K = g*16+e
            v16h bk = *(const v16h*)&sWt[2048 + col * 32 + g * 16];
            v16h bv = *(const v16h*)&sWt[2 * 2048 + col * 32 + g * 16];
            accq[nt] = wmma_f16(a.v, bq, accq[nt]);
            acck[nt] = wmma_f16(a.v, bk, acck[nt]);
            accv[nt] = wmma_f16(a.v, bv, accv[nt]);
        }
        __syncthreads();
    }

    // ---- RoPE on q,k (in-register, fp32) + store f16 ----
    // C/D layout: lane holds col n = nt*16+lane16; VGPR r holds row r + 8*g.
    // Pair (2i, 2i+1) lives in lanes (2j, 2j+1) -> __shfl_xor(...,1).
#pragma unroll
    for (int nt = 0; nt < 4; ++nt) {
        int n = nt * 16 + lane16;
        int i = n >> 1;
        float theta = __powf(10000.f, -2.f * ((float)i - 1.f) / 32.f);
#pragma unroll
        for (int r = 0; r < 8; ++r) {
            int m = m0 + r + 8 * g;
            int t = m & (Tc - 1);
            float sn, cs;
            __sincosf((float)t * theta, &sn, &cs);
            float qv = accq[nt][r];
            float kv = acck[nt][r];
            float qp = __shfl_xor(qv, 1, 32);
            float kp = __shfl_xor(kv, 1, 32);
            float qo, ko;
            if (n & 1) { qo = qv * cs - qp * sn; ko = kv * cs - kp * sn; }
            else       { qo = qv * cs + qp * sn; ko = kv * cs + kp * sn; }
            size_t ro = (size_t)m * Hc + n;
            qbuf[ro] = (_Float16)qo;
            kbuf[ro] = (_Float16)ko;
            int b = m >> 11;
            vT[(((size_t)(b * Hc + n)) << 11) + t] = (_Float16)accv[nt][r];
        }
    }
}

// ============================================================================
// Kernel 2: causal flash attention.
// Block: 8 waves x 16 query rows. K/V tiles (32 keys) staged once per block
// into LDS with async global->LDS copies, double-buffered over the kv loop.
// ============================================================================
__global__ __launch_bounds__(256) void attn_kernel(
    const _Float16* __restrict__ qbuf, const _Float16* __restrict__ kbuf,
    const _Float16* __restrict__ vT, float* __restrict__ out)
{
    __shared__ __align__(64) _Float16 sK[2][32 * 64];  // [tok][h] per tile
    __shared__ __align__(64) _Float16 sV[2][64 * 32];  // [h][tok] per tile
    __shared__ __align__(64) _Float16 sP[8 * 16 * 32]; // per-wave P tiles

    const int tid    = threadIdx.x;
    const int wave   = tid >> 5;
    const int lane   = tid & 31;
    const int lane16 = lane & 15;
    const int g      = lane >> 4;
    const int m0b    = blockIdx.x * 128;
    const int b      = m0b >> 11;
    const int qtb    = m0b & (Tc - 1);
    const int qt     = qtb + wave * 16;                 // wave's query token base
    const int mq0    = m0b + wave * 16;
    _Float16* myP    = sP + wave * 512;

    const int ntiles = (qt + 16 + 31) >> 5;             // this wave's causal tiles
    const int ntmax  = (qtb + 128) >> 5;                // tiles staged block-wide

    // per-thread async-copy addressing (16B per thread; 4KB per tile)
    const char* kgbase = (const char*)(kbuf + (size_t)b * Tc * Hc);
    const char* vgbase = (const char*)(vT + ((size_t)b * Hc << 11));
    const int vrow = tid >> 2, vchunk = tid & 3;
    const uint32_t dK[2] = { lds_off(&sK[0][0]) + (uint32_t)tid * 16,
                             lds_off(&sK[1][0]) + (uint32_t)tid * 16 };
    const uint32_t dV[2] = { lds_off(&sV[0][0]) + (uint32_t)(vrow * 64 + vchunk * 16),
                             lds_off(&sV[1][0]) + (uint32_t)(vrow * 64 + vchunk * 16) };

    // prologue: stage tile 0 into buffer 0
    async_b128(dK[0], kgbase + (size_t)tid * 16);
    async_b128(dV[0], vgbase + (((size_t)vrow << 11) + vchunk * 8) * 2);

    // Q A-fragments (K=0..31 and K=32..63), loaded once
    AFrag aq0, aq1;
    {
        const _Float16* p = qbuf + (size_t)(mq0 + lane16) * Hc;
        aq0.h8[0] = *(const v8h*)(p + g * 16);
        aq0.h8[1] = *(const v8h*)(p + 8 + g * 16);
        aq1.h8[0] = *(const v8h*)(p + 32 + g * 16);
        aq1.h8[1] = *(const v8h*)(p + 40 + g * 16);
    }

    v8f o[4] = {};
    float mrow[8], lrow[8];
#pragma unroll
    for (int r = 0; r < 8; ++r) { mrow[r] = -1e30f; lrow[r] = 0.f; }

    for (int jt = 0; jt < ntmax; ++jt) {
        const int j = jt << 5;
        const _Float16* kb = sK[jt & 1];
        const _Float16* vb = sV[jt & 1];

        // issue next tile's async copy into the other buffer, then wait for
        // the current tile (async ops complete in order -> cnt<=2 means all
        // earlier copies have landed in LDS).
        if (jt + 1 < ntmax) {
            const int jn = j + 32;
            async_b128(dK[(jt + 1) & 1], kgbase + (size_t)jn * 128 + tid * 16);
            async_b128(dV[(jt + 1) & 1],
                       vgbase + (((size_t)vrow << 11) + jn + vchunk * 8) * 2);
            asm volatile("s_wait_asynccnt 0x2" ::: "memory");
        } else {
            asm volatile("s_wait_asynccnt 0x0" ::: "memory");
        }
        __syncthreads();

        if (jt < ntiles) {
            // ---- scores: S[16x32] = Q[16x64] . K^T, 4 WMMAs ----
            v8f s0 = {}, s1 = {};
            v16h bk0 = *(const v16h*)&kb[lane16 * 64 + g * 16];
            v16h bk1 = *(const v16h*)&kb[lane16 * 64 + 32 + g * 16];
            s0 = wmma_f16(aq0.v, bk0, s0);
            s0 = wmma_f16(aq1.v, bk1, s0);
            v16h bk2 = *(const v16h*)&kb[(16 + lane16) * 64 + g * 16];
            v16h bk3 = *(const v16h*)&kb[(16 + lane16) * 64 + 32 + g * 16];
            s1 = wmma_f16(aq0.v, bk2, s1);
            s1 = wmma_f16(aq1.v, bk3, s1);

            // ---- online softmax (row = r + 8g, 16 lanes per row group) ----
            const int kc0 = j + lane16, kc1 = kc0 + 16;
#pragma unroll
            for (int r = 0; r < 8; ++r) {
                int rowtok = qt + r + 8 * g;
                float a0 = (kc0 <= rowtok) ? s0[r] * 0.125f : -1e30f;
                float a1 = (kc1 <= rowtok) ? s1[r] * 0.125f : -1e30f;
                float mx = fmaxf(a0, a1);
                mx = fmaxf(mx, __shfl_xor(mx, 1, 32));
                mx = fmaxf(mx, __shfl_xor(mx, 2, 32));
                mx = fmaxf(mx, __shfl_xor(mx, 4, 32));
                mx = fmaxf(mx, __shfl_xor(mx, 8, 32));
                float nm    = fmaxf(mrow[r], mx);
                float alpha = __expf(mrow[r] - nm);
                float p0    = __expf(a0 - nm);
                float p1    = __expf(a1 - nm);
                float rs = p0 + p1;
                rs += __shfl_xor(rs, 1, 32);
                rs += __shfl_xor(rs, 2, 32);
                rs += __shfl_xor(rs, 4, 32);
                rs += __shfl_xor(rs, 8, 32);
                lrow[r] = lrow[r] * alpha + rs;
                mrow[r] = nm;
#pragma unroll
                for (int nt = 0; nt < 4; ++nt) o[nt][r] *= alpha;
                // P tile row-major 16x32 (per-wave slice; DS ops in-order per wave)
                myP[(r + 8 * g) * 32 + lane16]      = (_Float16)p0;
                myP[(r + 8 * g) * 32 + 16 + lane16] = (_Float16)p1;
            }

            // ---- P A-fragment from LDS ----
            AFrag ap;
            ap.h8[0] = *(const v8h*)&myP[lane16 * 32 + g * 16];
            ap.h8[1] = *(const v8h*)&myP[lane16 * 32 + 8 + g * 16];

            // ---- O += P . V  (transposed V tile: contiguous B-fragments) ----
#pragma unroll
            for (int nt = 0; nt < 4; ++nt) {
                v16h bv = *(const v16h*)&vb[(nt * 16 + lane16) * 32 + g * 16];
                o[nt] = wmma_f16(ap.v, bv, o[nt]);
            }
        }
        __syncthreads();   // all waves done reading buf[jt&1] before overwrite
    }

    // ---- epilogue: divide by row sum, store fp32 ----
#pragma unroll
    for (int nt = 0; nt < 4; ++nt) {
        int n = nt * 16 + lane16;
#pragma unroll
        for (int r = 0; r < 8; ++r) {
            int rowtok = qt + r + 8 * g;
            out[(size_t)(b * Tc + rowtok) * Hc + n] = o[nt][r] / lrow[r];
        }
    }
}

// ============================================================================
extern "C" void kernel_launch(void* const* d_in, const int* in_sizes, int n_in,
                              void* d_out, int out_size, void* d_ws, size_t ws_size,
                              hipStream_t stream) {
    const float* x  = (const float*)d_in[0];
    const float* Wq = (const float*)d_in[1];
    const float* Wk = (const float*)d_in[2];
    const float* Wv = (const float*)d_in[3];
    float* out = (float*)d_out;

    _Float16* ws = (_Float16*)d_ws;
    _Float16* qb = ws;                                  // [BT,64] f16
    _Float16* kb = ws + (size_t)BT * Hc;                // [BT,64] f16
    _Float16* vt = ws + (size_t)2 * BT * Hc;            // [B,64,T] f16

    qkv_rope_kernel<<<BT / 128, 256, 0, stream>>>(x, Wq, Wk, Wv, qb, kb, vt);
    attn_kernel<<<BT / 128, 256, 0, stream>>>(qb, kb, vt, out);
}
